// MSDeformableAttention_45320494907560
// MI455X (gfx1250) — compile-verified, matrix operand order checked by reading, not run on hardware
//
#include <hip/hip_runtime.h>
#include <hip/hip_bf16.h>

typedef __attribute__((ext_vector_type(16))) _Float16 v16h;
typedef __attribute__((ext_vector_type(8)))  _Float16 v8h;
typedef __attribute__((ext_vector_type(8)))  float    v8f;

#define HH    8
#define LV    4
#define PN    4
#define CDIM  256
#define BATCH 2
#define NQ    10000

// ---------------- WMMA helper ----------------

__device__ __forceinline__ v8f wmma16(v16h a, v16h b, v8f c) {
  // D = A(16x32 f16) * B(32x16 f16) + C(16x16 f32)
  return __builtin_amdgcn_wmma_f32_16x16x32_f16(
      /*neg_a=*/false, a, /*neg_b=*/false, b,
      /*c_mod=*/(short)0, c, /*reuse_a=*/false, /*reuse_b=*/false);
}

__device__ __forceinline__ v16h load_a_frag(const _Float16* __restrict__ arow,
                                            int kt32, int g) {
  v8h alo = *(const v8h*)(arow + kt32 * 32 + g * 8);
  v8h ahi = *(const v8h*)(arow + kt32 * 32 + 16 + g * 8);
  v16h a;
#pragma unroll
  for (int j = 0; j < 8; ++j) { a[j] = alo[j]; a[8 + j] = ahi[j]; }
  return a;
}

// ---------------- Pack kernels (one-shot f32 -> f16 conversions) ----------------

// Row-major f32 -> f16, 4 elements per thread (query).
__global__ void cvt_f16_kernel(const float* __restrict__ in, _Float16* __restrict__ out,
                               int n4) {
  int tid = blockIdx.x * blockDim.x + threadIdx.x;
  if (tid >= n4) return;
  float4 v = ((const float4*)in)[tid];
  _Float16* o = out + (size_t)tid * 4;
  o[0] = (_Float16)v.x; o[1] = (_Float16)v.y;
  o[2] = (_Float16)v.z; o[3] = (_Float16)v.w;
}

// Weight K x N (f32, row-major) -> per-lane WMMA B-fragment layout:
//   out[((nt*(K/32) + kt32)*32 + lane)*16 + j] ,
//   j -> k = kt32*32 + (j/8)*16 + 8*(lane/16) + j%8 , n = nt*16 + lane%16
__global__ void packW_kernel(const float* __restrict__ W, _Float16* __restrict__ out, int N) {
  const int K = CDIM;
  int tid = blockIdx.x * blockDim.x + threadIdx.x;
  int total = (K / 32) * (N / 16) * 32;
  if (tid >= total) return;
  int lane = tid & 31;
  int bi = tid >> 5;
  int kt32 = bi % (K / 32);
  int nt = bi / (K / 32);
  int g = lane >> 4, r = lane & 15;
  int n = nt * 16 + r;
  int kbase = kt32 * 32 + g * 8;
  _Float16* o = out + (size_t)tid * 16;
#pragma unroll
  for (int j = 0; j < 8; ++j) {
    o[j]     = (_Float16)W[(size_t)(kbase + j) * N + n];
    o[8 + j] = (_Float16)W[(size_t)(kbase + 16 + j) * N + n];
  }
}

// NCHW value tensor -> channel-last f16 rows: out[(b*HW+s)*256 + c] = V[b][c][s]
__global__ void pack_value_kernel(const float* __restrict__ V, _Float16* __restrict__ out,
                                  int HW, int M) {
  size_t tid = (size_t)blockIdx.x * blockDim.x + threadIdx.x;
  size_t total = (size_t)M * CDIM;
  if (tid >= total) return;
  int c = (int)(tid & (CDIM - 1));
  int m = (int)(tid >> 8);
  int b = m / HW, s = m % HW;
  out[tid] = (_Float16)V[(size_t)b * CDIM * HW + (size_t)c * HW + s];
}

// ---------------- f16 GEMM, register-blocked 16x64 per wave, double-buffered ----------------
// C(f32) = A(f16, MxK=256 row-major) @ Bp(f16, packed fragments) + bias.
// Caller guarantees C rows writable up to ceil(M/16)*16.

__global__ void gemm_f16_kernel(const _Float16* __restrict__ A,
                                const _Float16* __restrict__ Bp,
                                const float* __restrict__ bias, float* __restrict__ C,
                                int M, int N) {
  const int K = CDIM;
  const int KB = K / 32;  // 8 k-blocks
  int wave = blockIdx.x * (blockDim.x >> 5) + (threadIdx.x >> 5);
  int lane = threadIdx.x & 31;
  int Mt = (M + 15) >> 4;
  int Ng = N >> 6;  // 64-column groups
  if (wave >= Mt * Ng) return;
  int mt = wave / Ng, ng = wave % Ng;
  int g = lane >> 4, r = lane & 15;
  int nb = ng * 64 + r;

  v8f c0, c1, c2, c3;
  {
    float b0 = bias[nb], b1 = bias[nb + 16], b2 = bias[nb + 32], b3 = bias[nb + 48];
#pragma unroll
    for (int i = 0; i < 8; ++i) { c0[i] = b0; c1[i] = b1; c2[i] = b2; c3[i] = b3; }
  }

  int m = mt * 16 + r;
  if (m >= M) m = M - 1;  // clamp (partial tile); padded/valid rows absorb stores
  const _Float16* arow = A + (size_t)m * K;
  const _Float16* bp = Bp + (size_t)(ng * 4) * KB * 512 + (size_t)lane * 16;

  // software pipeline: fragments for step kt32+1 are loaded (distinct registers)
  // before the WMMAs of step kt32 issue, so waits can be partial.
  v16h a_c  = load_a_frag(arow, 0, g);
  v16h b0_c = *(const v16h*)(bp);
  v16h b1_c = *(const v16h*)(bp + (size_t)KB * 512);
  v16h b2_c = *(const v16h*)(bp + (size_t)(2 * KB) * 512);
  v16h b3_c = *(const v16h*)(bp + (size_t)(3 * KB) * 512);

#pragma unroll
  for (int kt32 = 0; kt32 < KB; ++kt32) {
    v16h a_n, b0_n, b1_n, b2_n, b3_n;
    if (kt32 + 1 < KB) {
      a_n  = load_a_frag(arow, kt32 + 1, g);
      b0_n = *(const v16h*)(bp + (size_t)(kt32 + 1) * 512);
      b1_n = *(const v16h*)(bp + (size_t)(KB + kt32 + 1) * 512);
      b2_n = *(const v16h*)(bp + (size_t)(2 * KB + kt32 + 1) * 512);
      b3_n = *(const v16h*)(bp + (size_t)(3 * KB + kt32 + 1) * 512);
    }
    c0 = wmma16(a_c, b0_c, c0);
    c1 = wmma16(a_c, b1_c, c1);
    c2 = wmma16(a_c, b2_c, c2);
    c3 = wmma16(a_c, b3_c, c3);
    if (kt32 + 1 < KB) {
      a_c = a_n; b0_c = b0_n; b1_c = b1_n; b2_c = b2_n; b3_c = b3_n;
    }
  }

  float* crow = C + (size_t)(mt * 16 + g * 8) * N + nb;
#pragma unroll
  for (int rr = 0; rr < 8; ++rr) {
    float* q = crow + (size_t)rr * N;
    q[0]  = c0[rr];
    q[16] = c1[rr];
    q[32] = c2[rr];
    q[48] = c3[rr];
  }
}

// ---------------- Softmax over rows of 16 (levels*points), 2 rows per wave32 ----------------

__global__ void softmax16_kernel(float* __restrict__ attn, int nrows) {
  int wave = blockIdx.x * (blockDim.x >> 5) + (threadIdx.x >> 5);
  int lane = threadIdx.x & 31;
  int row = wave * 2 + (lane >> 4);
  if (row >= nrows) return;
  int e = lane & 15;
  float* p = attn + (size_t)row * 16 + e;
  float x = *p;
  float mx = x;
#pragma unroll
  for (int msk = 8; msk >= 1; msk >>= 1) mx = fmaxf(mx, __shfl_xor(mx, msk, 32));
  float ex = __expf(x - mx);
  float sm = ex;
#pragma unroll
  for (int msk = 8; msk >= 1; msk >>= 1) sm += __shfl_xor(sm, msk, 32);
  *p = ex / sm;
}

// ---------------- Bilinear sampling + attention-weighted aggregation ----------------
// one wave per (b, q, head); lane = channel within head (head dim 32 == wave32).
// Accumulates in f32, stores f16 rows directly consumable by the final WMMA GEMM.

__global__ void sample_kernel(const float* __restrict__ ref, const float* __restrict__ off,
                              const float* __restrict__ attn, const float* __restrict__ pv,
                              _Float16* __restrict__ acc) {
  const int    Hs[LV]  = {100, 50, 25, 13};
  const int    Ws[LV]  = {100, 50, 25, 13};
  const int    HWs[LV] = {10000, 2500, 625, 169};
  const size_t pvb[LV] = {0, 5120000, 6402048, 6725632};  // padded level bases (f32 elems)

  int wave = blockIdx.x * (blockDim.x >> 5) + (threadIdx.x >> 5);
  if (wave >= BATCH * NQ * HH) return;
  int lane = threadIdx.x & 31;
  int head = wave % HH;
  int bq   = wave / HH;
  int b    = bq / NQ;

  float rx = ref[(size_t)bq * 2 + 0];
  float ry = ref[(size_t)bq * 2 + 1];
  const float* offq = off  + (size_t)bq * (HH * LV * PN * 2) + head * (LV * PN * 2);
  const float* attq = attn + (size_t)bq * (HH * LV * PN)     + head * (LV * PN);

  float accv = 0.f;
#pragma unroll
  for (int lvl = 0; lvl < LV; ++lvl) {
    int W = Ws[lvl], H = Hs[lvl];
    const float* pvl = pv + pvb[lvl] + (size_t)b * HWs[lvl] * CDIM + head * 32 + lane;
#pragma unroll
    for (int p = 0; p < PN; ++p) {
      float lx = rx + offq[lvl * (PN * 2) + p * 2 + 0];
      float ly = ry + offq[lvl * (PN * 2) + p * 2 + 1];
      float aw = attq[lvl * PN + p];
      // grid = 2*loc-1; unnormalized (align_corners=False): x = loc*W - 0.5
      float x = lx * (float)W - 0.5f;
      float y = ly * (float)H - 0.5f;
      float xf = floorf(x), yf = floorf(y);
      int x0 = (int)xf, y0 = (int)yf;
      int x1 = x0 + 1,  y1 = y0 + 1;
      float wx1 = x - xf, wy1 = y - yf;
      float wx0 = 1.f - wx1, wy0 = 1.f - wy1;
      float v = 0.f;
      if (y0 >= 0 && y0 < H) {
        if (x0 >= 0 && x0 < W) v += wx0 * wy0 * pvl[(size_t)(y0 * W + x0) * CDIM];
        if (x1 >= 0 && x1 < W) v += wx1 * wy0 * pvl[(size_t)(y0 * W + x1) * CDIM];
      }
      if (y1 >= 0 && y1 < H) {
        if (x0 >= 0 && x0 < W) v += wx0 * wy1 * pvl[(size_t)(y1 * W + x0) * CDIM];
        if (x1 >= 0 && x1 < W) v += wx1 * wy1 * pvl[(size_t)(y1 * W + x1) * CDIM];
      }
      accv += aw * v;
    }
  }
  acc[(size_t)bq * CDIM + head * 32 + lane] = (_Float16)accv;
}

// ---------------- Host launch ----------------

extern "C" void kernel_launch(void* const* d_in, const int* in_sizes, int n_in,
                              void* d_out, int out_size, void* d_ws, size_t ws_size,
                              hipStream_t stream) {
  const float* query = (const float*)d_in[0];
  const float* ref   = (const float*)d_in[1];
  const float* val[LV] = {(const float*)d_in[2], (const float*)d_in[3],
                          (const float*)d_in[4], (const float*)d_in[5]};
  const float* Wv    = (const float*)d_in[6];
  const float* bv    = (const float*)d_in[7];
  const float* Woff  = (const float*)d_in[8];
  const float* boff  = (const float*)d_in[9];
  const float* Wattn = (const float*)d_in[10];
  const float* battn = (const float*)d_in[11];
  const float* Wout  = (const float*)d_in[12];
  const float* bout  = (const float*)d_in[13];
  float* out = (float*)d_out;

  // ---- workspace carve-up ----
  // f32 region (floats):  off 5,120,000 | attn 2,560,000 | pv(padded) 6,815,744
  // f16 region (halves):  qf16 5,120,000 | accf16 5,120,000 | vf16 6,815,744 | packed W 229,376
  float* ws    = (float*)d_ws;
  float* off   = ws;
  float* attn  = ws + 5120000;
  float* pv    = ws + 7680000;
  _Float16* hreg   = (_Float16*)(ws + 14495744);
  _Float16* qf16   = hreg;
  _Float16* accf16 = hreg + 5120000;
  _Float16* vf16   = hreg + 10240000;
  _Float16* Woffp  = hreg + 17055744;
  _Float16* Wattnp = Woffp + 65536;
  _Float16* Wvp    = Woffp + 98304;
  _Float16* Woutp  = Woffp + 163840;

  const int M = BATCH * NQ;               // 20000 (multiple of 16)
  const int HWs[LV]    = {10000, 2500, 625, 169};
  const size_t pvb[LV] = {0, 5120000, 6402048, 6725632};  // padded bases (elements)

  dim3 blk(128);   // 4 waves / block (GEMM)
  dim3 blkE(256);  // elementwise / softmax / sampling

  // 0) one-shot packs: weights -> fragment layout, query -> f16, values -> channel-last f16
  packW_kernel<<<(8 * 16 * 32 + 255) / 256, blkE, 0, stream>>>(Woff, Woffp, 256);
  packW_kernel<<<(8 * 8 * 32 + 255) / 256, blkE, 0, stream>>>(Wattn, Wattnp, 128);
  packW_kernel<<<(8 * 16 * 32 + 255) / 256, blkE, 0, stream>>>(Wv, Wvp, 256);
  packW_kernel<<<(8 * 16 * 32 + 255) / 256, blkE, 0, stream>>>(Wout, Woutp, 256);
  cvt_f16_kernel<<<(M * CDIM / 4 + 255) / 256, blkE, 0, stream>>>(query, qf16, M * CDIM / 4);
  for (int l = 0; l < LV; ++l) {
    int Ml = BATCH * HWs[l];
    size_t total = (size_t)Ml * CDIM;
    pack_value_kernel<<<(int)((total + 255) / 256), blkE, 0, stream>>>(val[l], vf16 + pvb[l],
                                                                       HWs[l], Ml);
  }

  // 1) sampling-offset projection: off = q @ Woff + boff   (20000x256x256)
  {
    int waves = (M / 16) * (CDIM / 64);
    gemm_f16_kernel<<<(waves + 3) / 4, blk, 0, stream>>>(qf16, Woffp, boff, off, M, CDIM);
  }
  // 2) attention-logit projection: attn = q @ Wattn + battn (20000x128x256)
  {
    int waves = (M / 16) * (128 / 64);
    gemm_f16_kernel<<<(waves + 3) / 4, blk, 0, stream>>>(qf16, Wattnp, battn, attn, M, 128);
  }
  // 3) softmax over levels*points (rows of 16); 160000 rows, 2 per wave
  {
    int nrows = M * HH;
    int waves = (nrows + 1) / 2;
    softmax16_kernel<<<(waves + 7) / 8, blkE, 0, stream>>>(attn, nrows);
  }
  // 4) value projection per level: pv = vf16 @ Wv + bv  (channel-last, padded scratch)
  for (int l = 0; l < LV; ++l) {
    int Ml = BATCH * HWs[l];
    int waves = ((Ml + 15) / 16) * (CDIM / 64);
    gemm_f16_kernel<<<(waves + 3) / 4, blk, 0, stream>>>(vf16 + pvb[l], Wvp, bv, pv + pvb[l],
                                                         Ml, CDIM);
  }
  // 5) bilinear gather + attention-weighted accumulation -> f16 acc rows
  {
    int waves = M * HH;  // 160000
    sample_kernel<<<(waves + 7) / 8, blkE, 0, stream>>>(ref, off, attn, pv, accf16);
  }
  // 6) output projection: out = acc @ Wout + bout
  {
    int waves = (M / 16) * (CDIM / 64);
    gemm_f16_kernel<<<(waves + 3) / 4, blk, 0, stream>>>(accf16, Woutp, bout, out, M, CDIM);
  }
}
